// CompletionNet_42786464203174
// MI455X (gfx1250) — matrix operand order repaired and checked by reading, not compiled, not run on hardware
//
#include <hip/hip_runtime.h>
#include <hip/hip_bf16.h>
#include <stdint.h>

// ---------------------------------------------------------------------------
// CDNA5 (gfx1250) sparse-completion U-net.
// Channel GEMMs run as implicit-GEMM on v_wmma_f32_16x16x32_bf16 (wave32).
// Each wave owns a 64(M)x16(N) tile (4 accumulators); tile coordinates are
// encoded in the launch grid (x = M-tile group, y = N-tile, z = tconv child
// parity) so there is no integer division in the kernel.  The 8 waves of a
// block share one N-tile, so their (pre-packed bf16) B-fragment stream is
// identical and L0-resident.  Weights are pre-packed once per call into
// WMMA-native fragments (2 x b128 per chunk); activations keep a bf16 mirror
// written by the fused BN+ELU+mask epilogues, so the inner loop does no
// format conversion at all.  Whole net is L2-resident on MI455X (192MB L2).
// ---------------------------------------------------------------------------

typedef __attribute__((ext_vector_type(16))) __bf16 v16bf;
typedef __attribute__((ext_vector_type(8)))  float  v8f;

struct alignas(16) q16 { unsigned int u[4]; };   // POD 16-byte block

union U16frag { v16bf v; __bf16 b[16]; unsigned short h[16]; };

__device__ __forceinline__ float elu1(float v) { return v > 0.f ? v : expm1f(v); }

// input voxel for (tap, output-row voxel); returns validity
__device__ __forceinline__ bool tapVox(int mode, int Din, int t,
                                       int rb, int rz, int ry, int rx,
                                       bool rvalid, size_t& vin) {
  int iz, iy, ix; bool ok = rvalid;
  if (mode == 0) {            // k3 s1 pad1
    iz = rz + t / 9 - 1; iy = ry + (t / 3) % 3 - 1; ix = rx + t % 3 - 1;
    ok = ok && (unsigned)iz < (unsigned)Din && (unsigned)iy < (unsigned)Din &&
               (unsigned)ix < (unsigned)Din;
  } else if (mode == 1) {     // k2 s2 VALID (always in-bounds spatially)
    iz = 2 * rz + (t >> 2); iy = 2 * ry + ((t >> 1) & 1); ix = 2 * rx + (t & 1);
  } else {                    // tconv: row voxel IS the coarse input voxel
    iz = rz; iy = ry; ix = rx;
  }
  if (!ok) { vin = 0; return false; }
  vin = (((size_t)rb * Din + iz) * Din + iy) * Din + ix;
  return true;
}

struct CP {
  const __bf16* src;          // bf16 activation mirror (channel-contiguous)
  float*  dst32;              // optional f32 output (cls/skip consumers)
  __bf16* dst16;              // optional bf16 output (next conv consumer)
  const __bf16* wpk;          // packed weights [q][nt][lane][16] bf16
  const float* gam; const float* bet;
  const int* mask;
  int B, Din, Dlin, Dout, Dlout;
  int cin, cout, cinC;        // cinC = cin/32 (0 when cin==16)
  int mode;                   // 0=k3s1p1, 1=k2s2 down, 2=k2s2 transpose
  int maskParent;             // epilogue mask indexed by parent voxel (dec k3)
  int NvRows, Mt4, Nt;
};

// grid: x = M-tile group (8 waves/block), y = N-tile, z = child parity (tconv)
__global__ __launch_bounds__(256) void k_conv_wmma(CP p) {
  const int lane = (int)(threadIdx.x & 31);
  const int mt   = (int)(blockIdx.x * 8 + (threadIdx.x >> 5));
  if (mt >= p.Mt4) return;                        // wave-uniform exit
  const int nt   = (int)blockIdx.y;
  const int ctap = (int)blockIdx.z;               // 0 for non-tconv
  const int half = lane >> 4;
  const int lr = lane & 15;
  const int n = nt * 16 + lr;

  const int rowD  = (p.mode == 2) ? p.Din  : p.Dout;
  const int rowDl = (p.mode == 2) ? p.Dlin : p.Dlout;
  const int rowDm = rowD - 1;

  int rb[4], rz[4], ry[4], rx[4]; bool rv[4];
#pragma unroll
  for (int s = 0; s < 4; ++s) {
    int vr = (mt * 4 + s) * 16 + lr;
    rv[s] = vr < p.NvRows;
    int v = rv[s] ? vr : 0;
    rx[s] = v & rowDm; ry[s] = (v >> rowDl) & rowDm;
    rz[s] = (v >> (2 * rowDl)) & rowDm; rb[s] = v >> (3 * rowDl);
  }
  const int tz = (ctap >> 2) & 1, ty = (ctap >> 1) & 1, tx = ctap & 1;
  const int wtapT = ((1 - tz) << 2) | ((1 - ty) << 1) | (1 - tx); // no kernel flip

  v8f acc[4];
#pragma unroll
  for (int s = 0; s < 4; ++s)
#pragma unroll
    for (int e = 0; e < 8; ++e) acc[s][e] = 0.f;

  const size_t cinS = (size_t)p.cin, coutS = (size_t)p.cout;
  const size_t wChunkStride = (size_t)p.Nt * 32 * 16;   // bf16 elems per chunk
  const q16 z16 = {{0u, 0u, 0u, 0u}};

  if (p.cin >= 32) {
    const int taps = (p.mode == 0) ? 27 : ((p.mode == 1) ? 8 : 1);
    for (int t = 0; t < taps; ++t) {
      const __bf16* abase[4]; bool vld[4]; bool any = false;
#pragma unroll
      for (int s = 0; s < 4; ++s) {
        size_t vin;
        vld[s] = tapVox(p.mode, p.Din, t, rb[s], rz[s], ry[s], rx[s], rv[s], vin);
        abase[s] = p.src + vin * cinS + half * 8;
        any = any || vld[s];
      }
      if (__ballot(any ? 1 : 0) == 0) continue;   // wave-uniform tap skip
      const int wt = (p.mode == 2) ? wtapT : t;
      const __bf16* wp = p.wpk + (size_t)(wt * p.cinC) * wChunkStride +
                         ((size_t)nt * 32 + lane) * 16;
      for (int cb = 0; cb < p.cin; cb += 32) {
        U16frag bf;
        *(q16*)&bf.h[0] = *(const q16*)wp;        // K-local 0..15 | 16..31
        *(q16*)&bf.h[8] = *(const q16*)(wp + 8);  // (by lane half)
        wp += wChunkStride;
#pragma unroll
        for (int s = 0; s < 4; ++s) {
          U16frag af;
          const __bf16* sp = abase[s] + cb;
          *(q16*)&af.h[0] = vld[s] ? *(const q16*)sp        : z16; // K 0..15
          *(q16*)&af.h[8] = vld[s] ? *(const q16*)(sp + 16) : z16; // K 16..31
          acc[s] = __builtin_amdgcn_wmma_f32_16x16x32_bf16(
              false, af.v, false, bf.v, (short)0, acc[s], false, false);
        }
      }
    }
  } else {
    // cin == 16: two taps per K=32 chunk (modes 0 and 1 only)
    const int taps = (p.mode == 0) ? 27 : 8;
    const int nq = (taps + 1) >> 1;
    const __bf16* wp = p.wpk + ((size_t)nt * 32 + lane) * 16;
    for (int q = 0; q < nq; ++q, wp += wChunkStride) {
      const int tA = 2 * q, tB = 2 * q + 1;
      const bool hasB = tB < taps;
      size_t vinA[4], vinB[4]; bool vldA[4], vldB[4]; bool any = false;
#pragma unroll
      for (int s = 0; s < 4; ++s) {
        vldA[s] = tapVox(p.mode, p.Din, tA, rb[s], rz[s], ry[s], rx[s], rv[s], vinA[s]);
        vldB[s] = hasB && tapVox(p.mode, p.Din, tB, rb[s], rz[s], ry[s], rx[s], rv[s], vinB[s]);
        any = any || vldA[s] || vldB[s];
      }
      if (__ballot(any ? 1 : 0) == 0) continue;
      U16frag bf;
      *(q16*)&bf.h[0] = *(const q16*)wp;
      *(q16*)&bf.h[8] = *(const q16*)(wp + 8);
#pragma unroll
      for (int s = 0; s < 4; ++s) {
        U16frag af;
        const __bf16* spA = p.src + vinA[s] * 16 + half * 8;
        const __bf16* spB = p.src + vinB[s] * 16 + half * 8;
        *(q16*)&af.h[0] = vldA[s] ? *(const q16*)spA : z16;
        *(q16*)&af.h[8] = vldB[s] ? *(const q16*)spB : z16;
        acc[s] = __builtin_amdgcn_wmma_f32_16x16x32_bf16(
            false, af.v, false, bf.v, (short)0, acc[s], false, false);
      }
    }
  }

  // fused epilogue: BN-affine + ELU + mask, dual-precision scatter store
  const float gm = p.gam[n], bt = p.bet[n];
  const int Do = p.Dout;
#pragma unroll
  for (int s = 0; s < 4; ++s) {
#pragma unroll
    for (int j = 0; j < 8; ++j) {
      int vr = (mt * 4 + s) * 16 + half * 8 + j;   // C row = half*8+j
      if (vr >= p.NvRows) continue;
      int xq = vr & rowDm, yq = (vr >> rowDl) & rowDm;
      int zq = (vr >> (2 * rowDl)) & rowDm, bq = vr >> (3 * rowDl);
      size_t vout; int mval;
      if (p.mode == 2) {
        int zf = 2 * zq + tz, yf = 2 * yq + ty, xf = 2 * xq + tx;
        vout = (((size_t)bq * Do + zf) * Do + yf) * Do + xf;
        mval = p.mask[vr];                         // mu(fine) = mdec(coarse row)
      } else {
        vout = (size_t)vr;
        if (p.maskParent) {
          int Dc = Do >> 1;
          size_t vc = (((size_t)bq * Dc + (zq >> 1)) * Dc + (yq >> 1)) * Dc + (xq >> 1);
          mval = p.mask[vc];
        } else mval = p.mask[vr];
      }
      float val = mval ? elu1(acc[s][j] * gm + bt) : 0.f;
      if (p.dst32) p.dst32[vout * coutS + n] = val;
      if (p.dst16) p.dst16[vout * coutS + n] = (__bf16)val;
    }
  }
}

// ---- weight pre-pack into WMMA fragment order -----------------------------
// packed[q][nt][lane][j] (bf16), flattened-K chunking identical to the conv
// kernel's A-fragment enumeration for both cin>=32 and cin==16 paths.
// grid: x = chunk q, y covers Nt*32 threads (no division in kernel).
__global__ __launch_bounds__(256) void k_repack(
    const float* __restrict__ w, __bf16* __restrict__ wp,
    int taps, int cin, int cinLog, int cout, int Nt) {
  int q = (int)blockIdx.x;
  int t2 = (int)(blockIdx.y * 256 + threadIdx.x);
  if (t2 >= Nt * 32) return;
  int lane = t2 & 31, nt = t2 >> 5;
  int half = lane >> 4, lr = lane & 15;
  int n = nt * 16 + lr;
  int Ktot = taps * cin;
  __bf16* o = wp + ((size_t)q * Nt * 32 + t2) * 16;
#pragma unroll 4
  for (int j = 0; j < 16; ++j) {
    int K = q * 32 + ((j >> 3) << 4) + half * 8 + (j & 7);
    float v = 0.f;
    if (K < Ktot) {
      int t = K >> cinLog, c = K & (cin - 1);
      v = w[((size_t)t * cin + c) * cout + n];
    }
    o[j] = (__bf16)v;
  }
}

// ---- encoder block 0: k3 conv 1->16 (VALU; K too small for WMMA) ----------
__global__ __launch_bounds__(256) void k_conv0(
    const float* __restrict__ x, const int* __restrict__ occ,
    const float* __restrict__ w, const float* __restrict__ g,
    const float* __restrict__ b, float* __restrict__ dst,
    __bf16* __restrict__ dsth, int Bn) {
  int v = blockIdx.x * blockDim.x + threadIdx.x;
  int Nv = Bn << 18;
  if (v >= Nv) return;
  int xq = v & 63, yq = (v >> 6) & 63, zq = (v >> 12) & 63, bq = v >> 18;
  float acc[16];
#pragma unroll
  for (int n = 0; n < 16; ++n) acc[n] = 0.f;
  for (int t = 0; t < 27; ++t) {
    int iz = zq + t / 9 - 1, iy = yq + (t / 3) % 3 - 1, ix = xq + t % 3 - 1;
    if ((unsigned)iz < 64u && (unsigned)iy < 64u && (unsigned)ix < 64u) {
      size_t vi = (((size_t)bq * 64 + iz) * 64 + iy) * 64 + ix;
      float xv = occ[vi] ? x[vi] : 0.f;            // x * occ (sparse support)
#pragma unroll
      for (int n = 0; n < 16; ++n) acc[n] += xv * w[t * 16 + n];
    }
  }
  int m = occ[v];
#pragma unroll
  for (int n = 0; n < 16; ++n) {
    float val = m ? elu1(acc[n] * g[n] + b[n]) : 0.f;
    dst [(size_t)v * 16 + n] = val;
    dsth[(size_t)v * 16 + n] = (__bf16)val;
  }
}

// ---- 2x occupancy pooling (any of 8 children) -----------------------------
__global__ __launch_bounds__(256) void k_pool2(
    const int* __restrict__ in, int* __restrict__ out, int Bn, int Dc, int Dcl) {
  int v = blockIdx.x * blockDim.x + threadIdx.x;
  int Nv = Bn * Dc * Dc * Dc;
  if (v >= Nv) return;
  int Dm = Dc - 1;
  int xq = v & Dm, yq = (v >> Dcl) & Dm, zq = (v >> (2 * Dcl)) & Dm, bq = v >> (3 * Dcl);
  int Di = Dc * 2, r = 0;
#pragma unroll
  for (int t = 0; t < 8; ++t) {
    int iz = 2 * zq + (t >> 2), iy = 2 * yq + ((t >> 1) & 1), ix = 2 * xq + (t & 1);
    r |= in[(((size_t)bq * Di + iz) * Di + iy) * Di + ix];
  }
  out[v] = r ? 1 : 0;
}

// ---- skip-add + 1x1 classifier + target gather + dynamic pruning ----------
__global__ __launch_bounds__(256) void k_cls(
    const float* __restrict__ d2, const float* __restrict__ fsk,
    const float* __restrict__ cw, const int* __restrict__ mdecC,
    const int* __restrict__ msk, const int* __restrict__ tp,
    float* __restrict__ outC, float* __restrict__ outT,
    float* __restrict__ decN, __bf16* __restrict__ decNh,
    int* __restrict__ mdecN, int Bn, int Df, int Dfl, int cout) {
  int v = blockIdx.x * blockDim.x + threadIdx.x;
  int Nv = Bn * Df * Df * Df;
  if (v >= Nv) return;
  int Dm = Df - 1;
  int xq = v & Dm, yq = (v >> Dfl) & Dm, zq = (v >> (2 * Dfl)) & Dm, bq = v >> (3 * Dfl);
  int Dc = Df >> 1;
  size_t vc = (((size_t)bq * Dc + (zq >> 1)) * Dc + (yq >> 1)) * Dc + (xq >> 1);
  int mu = mdecC[vc];
  int un = (mu | msk[v]) ? 1 : 0;                  // union of coordinate sets
  const float* dp = d2  + (size_t)v * cout;
  const float* fp = fsk + (size_t)v * cout;
  float s = 0.f;
  for (int c = 0; c < cout; ++c) s += (dp[c] + fp[c]) * cw[c];
  int keep = (un && s > 0.f) ? 1 : 0;
  outC[v] = un ? s : 0.f;
  outT[v] = (un && tp[v]) ? 1.f : 0.f;
  mdecN[v] = keep;
  float km = keep ? 1.f : 0.f;
  for (int c = 0; c < cout; ++c) {
    float val = (dp[c] + fp[c]) * km;
    decN [(size_t)v * cout + c] = val;
    decNh[(size_t)v * cout + c] = (__bf16)val;
  }
}

// ---------------------------------------------------------------------------
static int log2i(int v) { int l = 0; while ((1 << l) < v) ++l; return l; }

extern "C" void kernel_launch(void* const* d_in, const int* in_sizes, int n_in,
                              void* d_out, int out_size, void* d_ws, size_t ws_size,
                              hipStream_t stream) {
  const int Bn = 2;
  const float* x   = (const float*)d_in[0];
  const int*   occ = (const int*)d_in[1];
  const int*   tgt = (const int*)d_in[2];

  // --- param pointers; support both plausible flatten orders --------------
  const float* clsW[6];
  const float *db1[6], *db2[6], *dg1[6], *dg2[6], *dw2[6], *dwt[6];
  const float *e0b, *e0g, *e0w;
  const float *ebd[7], *egd[7], *ewd[7], *ebk[7], *egk[7], *ewk[7];
  if (in_sizes[3] == 512) {
    // jax tree_flatten: sorted dict keys: cls, dec(b1,b2,g1,g2,w2,wt), enc(b,g,w)
    int k = 3;
    for (int i = 0; i < 6; ++i) clsW[i] = (const float*)d_in[k++];
    for (int i = 0; i < 6; ++i) {
      db1[i] = (const float*)d_in[k++]; db2[i] = (const float*)d_in[k++];
      dg1[i] = (const float*)d_in[k++]; dg2[i] = (const float*)d_in[k++];
      dw2[i] = (const float*)d_in[k++]; dwt[i] = (const float*)d_in[k++];
    }
    e0b = (const float*)d_in[k++]; e0g = (const float*)d_in[k++]; e0w = (const float*)d_in[k++];
    for (int b = 1; b <= 6; ++b) {
      ebd[b] = (const float*)d_in[k++]; egd[b] = (const float*)d_in[k++]; ewd[b] = (const float*)d_in[k++];
      ebk[b] = (const float*)d_in[k++]; egk[b] = (const float*)d_in[k++]; ewk[b] = (const float*)d_in[k++];
    }
  } else {
    // insertion order: enc(w,g,b per layer), dec(wt,g1,b1,w2,g2,b2), cls
    int k = 3;
    e0w = (const float*)d_in[k++]; e0g = (const float*)d_in[k++]; e0b = (const float*)d_in[k++];
    for (int b = 1; b <= 6; ++b) {
      ewd[b] = (const float*)d_in[k++]; egd[b] = (const float*)d_in[k++]; ebd[b] = (const float*)d_in[k++];
      ewk[b] = (const float*)d_in[k++]; egk[b] = (const float*)d_in[k++]; ebk[b] = (const float*)d_in[k++];
    }
    for (int i = 0; i < 6; ++i) {
      dwt[i] = (const float*)d_in[k++]; dg1[i] = (const float*)d_in[k++]; db1[i] = (const float*)d_in[k++];
      dw2[i] = (const float*)d_in[k++]; dg2[i] = (const float*)d_in[k++]; db2[i] = (const float*)d_in[k++];
    }
    for (int i = 0; i < 6; ++i) clsW[i] = (const float*)d_in[k++];
  }

  // --- workspace layout ----------------------------------------------------
  char* wsp = (char*)d_ws; size_t off = 0;
  auto alloc = [&](size_t bytes) -> void* {
    void* p = wsp + off; off = (off + bytes + 255) & ~(size_t)255; return p;
  };
  static const int Dg[7] = {64, 32, 16, 8, 4, 2, 1};
  static const int ch[7] = {16, 32, 64, 128, 256, 512, 1024};
  float*  feat [7];
  __bf16* featH[7];
  for (int k = 0; k < 7; ++k) {
    size_t n = (size_t)Bn * Dg[k] * Dg[k] * Dg[k] * ch[k];
    feat [k] = (float*) alloc(n * 4);
    featH[k] = (__bf16*)alloc(n * 2);
  }
  const int* msk[7]; msk[0] = occ;
  for (int k = 1; k < 7; ++k)
    msk[k] = (int*)alloc((size_t)Bn * Dg[k] * Dg[k] * Dg[k] * 4);
  int* tpb[5]; // pooled target at grids 32,16,8,4,2
  for (int k = 0; k < 5; ++k) {
    int g = 32 >> k;
    tpb[k] = (int*)alloc((size_t)Bn * g * g * g * 4);
  }
  __bf16* P1h = (__bf16*)alloc((size_t)8388608 * 2);  // tconv / down-conv out
  float*  P2  = (float*) alloc((size_t)8388608 * 4);  // decoder k3 out (f32)
  __bf16* P0h = (__bf16*)alloc((size_t)8388608 * 2);  // pruned dec (bf16 mirror)
  int* MA = (int*)alloc((size_t)524288 * 4);
  int* MB = (int*)alloc((size_t)524288 * 4);

  // --- output offsets (cls[0..5], tgt[0..5], dec) --------------------------
  float* out = (float*)d_out;
  size_t cls_off[6], tgt_off[6], o = 0;
  for (int i = 0; i < 6; ++i) { cls_off[i] = o; int G = 2 << i; o += (size_t)Bn * G * G * G; }
  for (int i = 0; i < 6; ++i) { tgt_off[i] = o; int G = 2 << i; o += (size_t)Bn * G * G * G; }
  size_t dec_off = o;
  float* P0 = out + dec_off;   // decoder f32 ping buffer == final dec slot

  auto conv = [&](const __bf16* src, float* dst32, __bf16* dst16,
                  const float* w, const float* g, const float* b,
                  const int* mask, int Din, int Dout,
                  int cin, int cout, int mode, int maskParent) {
    int tapsW = (mode == 0) ? 27 : 8;            // weight taps (tconv has 8)
    int Ktot = tapsW * cin;
    int nchunks = (Ktot + 31) / 32;
    int Nt = cout / 16;
    __bf16* wpk = (__bf16*)alloc((size_t)nchunks * Nt * 32 * 16 * 2);
    dim3 rg(nchunks, (Nt * 32 + 255) / 256);
    k_repack<<<rg, 256, 0, stream>>>(w, wpk, tapsW, cin, log2i(cin), cout, Nt);
    CP p; p.src = src; p.dst32 = dst32; p.dst16 = dst16; p.wpk = wpk;
    p.gam = g; p.bet = b; p.mask = mask;
    p.B = Bn; p.Din = Din; p.Dlin = log2i(Din); p.Dout = Dout; p.Dlout = log2i(Dout);
    p.cin = cin; p.cout = cout; p.cinC = cin >> 5;
    p.mode = mode; p.maskParent = maskParent;
    int rowD = (mode == 2) ? Din : Dout;
    p.NvRows = Bn * rowD * rowD * rowD;
    p.Mt4 = (p.NvRows + 63) / 64;
    p.Nt = Nt;
    dim3 cg((p.Mt4 + 7) / 8, Nt, (mode == 2) ? 8 : 1);
    k_conv_wmma<<<cg, 256, 0, stream>>>(p);
  };

  // --- mask + target pyramids ---------------------------------------------
  for (int k = 1; k < 7; ++k) {
    int Dc = Dg[k], Nv = Bn * Dc * Dc * Dc;
    k_pool2<<<(Nv + 255) / 256, 256, 0, stream>>>(msk[k - 1], (int*)msk[k], Bn, Dc, log2i(Dc));
  }
  for (int k = 0; k < 5; ++k) {
    int g = 32 >> k, Nv = Bn * g * g * g;
    const int* src = (k == 0) ? tgt : tpb[k - 1];
    k_pool2<<<(Nv + 255) / 256, 256, 0, stream>>>(src, tpb[k], Bn, g, log2i(g));
  }

  // --- encoder -------------------------------------------------------------
  {
    int Nv = Bn * 64 * 64 * 64;
    k_conv0<<<(Nv + 255) / 256, 256, 0, stream>>>(
        x, occ, e0w, e0g, e0b, feat[0], featH[0], Bn);
  }
  for (int b = 1; b <= 6; ++b) {
    conv(featH[b - 1], nullptr, P1h, ewd[b], egd[b], ebd[b], msk[b],
         Dg[b - 1], Dg[b], ch[b - 1], ch[b], 1, 0);            // k2 s2 down
    conv(P1h, feat[b], featH[b], ewk[b], egk[b], ebk[b], msk[b],
         Dg[b], Dg[b], ch[b], ch[b], 0, 0);                    // k3 s1
  }

  // --- decoder with dynamic pruning ---------------------------------------
  const __bf16* decCur = featH[6];
  const int* mdecCur = msk[6];
  for (int i = 0; i < 6; ++i) {
    int Dc = 1 << i, Df = 2 << i;
    int ci = 1024 >> i, co = 512 >> i;
    conv(decCur, nullptr, P1h, dwt[i], dg1[i], db1[i], mdecCur,
         Dc, Df, ci, co, 2, 0);                                // tconv k2 s2
    conv(P1h, P2, nullptr, dw2[i], dg2[i], db2[i], mdecCur,
         Df, Df, co, co, 0, 1);                                // k3 (mask=up2(mdec))
    const int* tpp = (Df == 64) ? tgt : tpb[4 - i];            // stride pyramid
    int Nvf = Bn * Df * Df * Df;
    int* mdecN = (i & 1) ? MB : MA;
    k_cls<<<(Nvf + 255) / 256, 256, 0, stream>>>(
        P2, feat[5 - i], clsW[i], mdecCur, msk[5 - i], tpp,
        out + cls_off[i], out + tgt_off[i], P0, P0h, mdecN,
        Bn, Df, log2i(Df), co);
    decCur = P0h; mdecCur = mdecN;
  }
  (void)n_in; (void)out_size; (void)ws_size; (void)in_sizes; (void)feat;
}
